// ShadowLoss_51110110822512
// MI455X (gfx1250) — matrix operand order repaired and checked by reading, not compiled, not run on hardware
//
#include <hip/hip_runtime.h>
#include <hip/hip_bf16.h>
#include <math.h>

// ---------------------------------------------------------------------------
// ShadowLoss fused kernel for gfx1250 (MI455X, CDNA5, wave32, WMMA)
//   B = 8192 rows, D = 128 features, TEMPERATURE = 0.1
// ---------------------------------------------------------------------------

#define BROWS 8192
#define DDIM  128
#define TILE  128            // workgroup tile (128x128 of the Gram matrix)
#define NTX   64             // grid dim = BROWS / TILE

typedef __attribute__((ext_vector_type(16))) _Float16 v16h;
typedef __attribute__((ext_vector_type(8)))  float    v8f;
typedef __attribute__((ext_vector_type(4)))  float    f32x4;

union FragU {
    v16h  h;
    f32x4 f[2];
};

// A-matrix 16x32 f16 fragment (ISA 7.12.2):
//   lanes 0-15 : M=lane,    v0-3 hold K=0..7,  v4-7 hold K=16..23
//   lanes 16-31: M=lane-16, v0-3 hold K=8..15, v4-7 hold K=24..31
// -> two 16-byte loads at (half*8) and (16 + half*8) halfwords.
__device__ __forceinline__ v16h load_fragA(const _Float16* __restrict__ p, int half) {
    FragU u;
    u.f[0] = *reinterpret_cast<const f32x4*>(p + half * 8);
    u.f[1] = *reinterpret_cast<const f32x4*>(p + 16 + half * 8);
    return u.h;
}

// B-matrix 32x16 f16 fragment (same K order as A):
//   lanes 0-15 : N=lane,    v0-3 hold K=0..7,  v4-7 hold K=16..23
//   lanes 16-31: N=lane-16, v0-3 hold K=8..15, v4-7 hold K=24..31
// -> two 16-byte loads at (half*8) and (16 + half*8) halfwords.
__device__ __forceinline__ v16h load_fragB(const _Float16* __restrict__ p, int half) {
    FragU u;
    u.f[0] = *reinterpret_cast<const f32x4*>(p + half * 8);
    u.f[1] = *reinterpret_cast<const f32x4*>(p + 16 + half * 8);
    return u.h;
}

// ---------------------------------------------------------------------------
// Kernel 1: L2-normalize rows -> f16, anchor[i] = sum(f_i^2), labels -> int32
// ---------------------------------------------------------------------------
__global__ __launch_bounds__(DDIM)
void shadow_normalize_kernel(const float* __restrict__ feat,
                             const int*   __restrict__ labels,
                             _Float16*    __restrict__ f16out,
                             float*       __restrict__ anchor,
                             int*         __restrict__ lab32) {
    const int row = blockIdx.x;
    const int t   = threadIdx.x;                 // 0..127, one per feature
    const float x = feat[(size_t)row * DDIM + t];

    float xs = x * x;
    // wave32 reduce
    #pragma unroll
    for (int off = 16; off > 0; off >>= 1)
        xs += __shfl_xor(xs, off, 32);

    __shared__ float wsum[4];
    if ((t & 31) == 0) wsum[t >> 5] = xs;
    __syncthreads();
    const float total = wsum[0] + wsum[1] + wsum[2] + wsum[3];

    const float norm = sqrtf(total);
    const float inv  = 1.0f / fmaxf(norm, 1e-12f);
    f16out[(size_t)row * DDIM + t] = (_Float16)(x * inv);

    if (t == 0) {
        anchor[row] = total * inv * inv;         // == sum(f*f) after normalize
        lab32[row]  = labels[row];
    }
}

// ---------------------------------------------------------------------------
// Kernel 2: fused Gram-matrix (WMMA f16 -> f32) + shadow loss epilogue.
// 8 waves per block; wave (wm,wn) owns a 32x64 subtile = 2x4 accum tiles.
// ---------------------------------------------------------------------------
__global__ __launch_bounds__(256)
void shadow_gram_loss_kernel(const _Float16* __restrict__ f,
                             const float*    __restrict__ anchor,
                             const int*      __restrict__ lab,
                             float*          __restrict__ partials) {
    const int tid  = threadIdx.x;
    const int wave = tid >> 5;
    const int lane = tid & 31;
    const int l16  = lane & 15;
    const int half = lane >> 4;

    const int tileM = blockIdx.x * TILE;
    const int tileN = blockIdx.y * TILE;
    const int wm = wave & 3;                  // 4 row groups of 32
    const int wn = wave >> 2;                 // 2 col groups of 64
    const int rowBase = tileM + wm * 32;
    const int colBase = tileN + wn * 64;

    const _Float16* __restrict__ pA0 = f + (size_t)(rowBase + l16) * DDIM;
    const _Float16* __restrict__ pA1 = f + (size_t)(rowBase + 16 + l16) * DDIM;
    const _Float16* __restrict__ pB0 = f + (size_t)(colBase +  0 + l16) * DDIM;
    const _Float16* __restrict__ pB1 = f + (size_t)(colBase + 16 + l16) * DDIM;
    const _Float16* __restrict__ pB2 = f + (size_t)(colBase + 32 + l16) * DDIM;
    const _Float16* __restrict__ pB3 = f + (size_t)(colBase + 48 + l16) * DDIM;

    v8f acc[2][4] = {};

    #pragma unroll
    for (int ks = 0; ks < DDIM; ks += 32) {
        const v16h a0 = load_fragA(pA0 + ks, half);
        const v16h a1 = load_fragA(pA1 + ks, half);
        const v16h b0 = load_fragB(pB0 + ks, half);
        const v16h b1 = load_fragB(pB1 + ks, half);
        const v16h b2 = load_fragB(pB2 + ks, half);
        const v16h b3 = load_fragB(pB3 + ks, half);

        acc[0][0] = __builtin_amdgcn_wmma_f32_16x16x32_f16(false, a0, false, b0, (short)0, acc[0][0], false, false);
        acc[0][1] = __builtin_amdgcn_wmma_f32_16x16x32_f16(false, a0, false, b1, (short)0, acc[0][1], false, false);
        acc[0][2] = __builtin_amdgcn_wmma_f32_16x16x32_f16(false, a0, false, b2, (short)0, acc[0][2], false, false);
        acc[0][3] = __builtin_amdgcn_wmma_f32_16x16x32_f16(false, a0, false, b3, (short)0, acc[0][3], false, false);
        acc[1][0] = __builtin_amdgcn_wmma_f32_16x16x32_f16(false, a1, false, b0, (short)0, acc[1][0], false, false);
        acc[1][1] = __builtin_amdgcn_wmma_f32_16x16x32_f16(false, a1, false, b1, (short)0, acc[1][1], false, false);
        acc[1][2] = __builtin_amdgcn_wmma_f32_16x16x32_f16(false, a1, false, b2, (short)0, acc[1][2], false, false);
        acc[1][3] = __builtin_amdgcn_wmma_f32_16x16x32_f16(false, a1, false, b3, (short)0, acc[1][3], false, false);
    }

    // ---- fused loss epilogue -------------------------------------------
    // C/D layout: lane (l16 -> col), VGPR e -> row e + 8*half within tile.
    int labN[4];
    #pragma unroll
    for (int nt = 0; nt < 4; ++nt)
        labN[nt] = lab[colBase + nt * 16 + l16];

    float sum = 0.0f;
    #pragma unroll
    for (int mt = 0; mt < 2; ++mt) {
        #pragma unroll
        for (int e = 0; e < 8; ++e) {
            const int   i  = rowBase + mt * 16 + half * 8 + e;
            const float ai = anchor[i];
            const int   li = lab[i];
            #pragma unroll
            for (int nt = 0; nt < 4; ++nt) {
                const int   j   = colBase + nt * 16 + l16;
                const float sim = acc[mt][nt][e];
                const float lg  = (ai - sim) * 10.0f;   // 1/TEMPERATURE
                const float v   = (li == labN[nt]) ? fmaxf(1.0f - lg, 0.0f)
                                                   : fmaxf(1.0f + lg, 0.0f);
                sum += (i != j) ? v : 0.0f;
            }
        }
    }

    // wave reduce then cross-wave reduce
    #pragma unroll
    for (int off = 16; off > 0; off >>= 1)
        sum += __shfl_xor(sum, off, 32);

    __shared__ float swave[8];
    if (lane == 0) swave[wave] = sum;
    __syncthreads();
    if (tid == 0) {
        float tot = 0.0f;
        #pragma unroll
        for (int w = 0; w < 8; ++w) tot += swave[w];
        partials[blockIdx.y * NTX + blockIdx.x] = tot;
    }
}

// ---------------------------------------------------------------------------
// Kernel 3: deterministic fixed-order reduction of 4096 block partials.
// ---------------------------------------------------------------------------
__global__ __launch_bounds__(256)
void shadow_reduce_kernel(const float* __restrict__ partials,
                          float*       __restrict__ out) {
    __shared__ float s[256];
    const int t = threadIdx.x;
    float sum = 0.0f;
    #pragma unroll
    for (int i = 0; i < NTX * NTX / 256; ++i)
        sum += partials[t + i * 256];
    s[t] = sum;
    __syncthreads();
    #pragma unroll
    for (int off = 128; off > 0; off >>= 1) {
        if (t < off) s[t] += s[t + off];
        __syncthreads();
    }
    if (t == 0)
        out[0] = s[0] * (1.0f / ((float)BROWS * (float)(BROWS - 1)));
}

// ---------------------------------------------------------------------------
// Launch
// ---------------------------------------------------------------------------
extern "C" void kernel_launch(void* const* d_in, const int* in_sizes, int n_in,
                              void* d_out, int out_size, void* d_ws, size_t ws_size,
                              hipStream_t stream) {
    const float* features = (const float*)d_in[0];
    const int*   labels   = (const int*)d_in[1];
    float*       out      = (float*)d_out;

    char* ws = (char*)d_ws;
    _Float16* f16   = (_Float16*)(ws);                              // 2 MB
    float*    anch  = (float*)(ws + (size_t)BROWS * DDIM * 2);      // 32 KB
    int*      lab32 = (int*)(ws + (size_t)BROWS * DDIM * 2 + BROWS * 4);
    float*    part  = (float*)(ws + (size_t)BROWS * DDIM * 2 + BROWS * 8);

    shadow_normalize_kernel<<<BROWS, DDIM, 0, stream>>>(features, labels, f16, anch, lab32);

    dim3 grid(NTX, NTX);
    shadow_gram_loss_kernel<<<grid, 256, 0, stream>>>(f16, anch, lab32, part);

    shadow_reduce_kernel<<<1, 256, 0, stream>>>(part, out);
}
